// MultiInnerHeadAttention_3504693313973
// MI455X (gfx1250) — matrix operand order repaired and checked by reading, not compiled, not run on hardware
//
#include <hip/hip_runtime.h>
#include <hip/hip_bf16.h>

// ---------------- problem constants ----------------
#define C_DIM 448
#define T_DIM 2048
#define B_DIM 4
#define H_DIM 8
#define D_HEAD 56
#define D_PAD 64
#define BT (B_DIM * T_DIM)            // 8192 tokens
#define BHT (B_DIM * H_DIM * T_DIM)   // 65536 (b,h,t) rows
#define SCALE_Q 0.047245559f          // 1/sqrt(448): reference scales by embed dim

typedef unsigned short u16;
typedef __bf16 bf16_t;
typedef bf16_t v16bf __attribute__((ext_vector_type(16)));
typedef float  v8f   __attribute__((ext_vector_type(8)));

union FragBF {
  v16bf v;
  u16 h[16];
  unsigned u[8];
};

__device__ __forceinline__ u16 f2bf(float x) {
  unsigned u = __builtin_bit_cast(unsigned, x);
  unsigned r = u + 0x7FFFu + ((u >> 16) & 1u);   // round-to-nearest-even
  return (u16)(r >> 16);
}

__device__ __forceinline__ v8f wmma_bf16(v16bf a, v16bf b, v8f c) {
  return __builtin_amdgcn_wmma_f32_16x16x32_bf16(
      false, a, false, b, (short)0, c, false, false);
}

// ---------------- async global->LDS staging (CDNA5 ASYNCcnt path) ----------------
#if defined(__has_builtin)
#  if __has_builtin(__builtin_amdgcn_global_load_async_to_lds_b128)
#    define HAVE_ASYNC_LDS 1
#  endif
#endif
#ifndef HAVE_ASYNC_LDS
#  define HAVE_ASYNC_LDS 0
#endif

#if HAVE_ASYNC_LDS
typedef int v4i_vs __attribute__((vector_size(16)));
typedef __attribute__((address_space(1))) v4i_vs* as1_v4i;
typedef __attribute__((address_space(3))) v4i_vs* as3_v4i;
#endif

__device__ __forceinline__ void cp16(const u16* g, u16* l) {
#if HAVE_ASYNC_LDS
  __builtin_amdgcn_global_load_async_to_lds_b128(
      (as1_v4i)(u16*)g, (as3_v4i)l, 0, 0);
#else
  *(uint4*)l = *(const uint4*)g;
#endif
}

__device__ __forceinline__ void cp_wait() {
#if HAVE_ASYNC_LDS
#  if __has_builtin(__builtin_amdgcn_s_wait_asynccnt)
  __builtin_amdgcn_s_wait_asynccnt(0);
#  else
  asm volatile("s_wait_asynccnt 0x0" ::: "memory");
#  endif
#endif
}

// A-matrix 16x32 bf16 fragment from row-major storage (row stride `mstride`).
// ISA layout: lane L -> M = L%16; dword j2 holds K=(j2<4?0:16)+(L/16)*8+2*(j2%4).
__device__ __forceinline__ v16bf load_fragA_rm(const u16* base, int mstride, int koff) {
  int lane = (int)(threadIdx.x & 31);
  int hi = lane >> 4;
  const u16* p = base + (size_t)(lane & 15) * mstride;
  FragBF f;
#pragma unroll
  for (int j2 = 0; j2 < 8; ++j2) {
    int k = ((j2 < 4) ? 0 : 16) + hi * 8 + (j2 & 3) * 2 + koff;
    f.u[j2] = *(const unsigned*)(p + k);
  }
  return f.v;
}

// B-matrix 32x16 bf16 fragment where K is contiguous in memory:
// element (k, n) at base[n*nstride + k]. Lane L -> N=L%16; dword j2 holds K=(j2<4?0:16)+(L/16)*8+2*(j2%4).
__device__ __forceinline__ v16bf load_fragB_kcontig(const u16* base, int nstride, int koff) {
  int lane = (int)(threadIdx.x & 31);
  int hi = lane >> 4;
  const u16* p = base + (size_t)(lane & 15) * nstride;
  FragBF f;
#pragma unroll
  for (int j2 = 0; j2 < 8; ++j2) {
    int k = ((j2 < 4) ? 0 : 16) + hi * 8 + (j2 & 3) * 2 + koff;
    f.u[j2] = *(const unsigned*)(p + k);
  }
  return f.v;
}

// ---------------- kernel 1a: f32 -> bf16 convert (x) ----------------
__global__ void k_f32_to_bf16(const float* __restrict__ src, u16* __restrict__ dst, int n) {
  int i = blockIdx.x * blockDim.x + threadIdx.x;
  int stride = gridDim.x * blockDim.x;
  for (; i < n; i += stride) dst[i] = f2bf(src[i]);
}

// ---------------- kernel 1b: weight f32 [K][N] -> bf16 transposed [N][K] ----------
__global__ void k_w_to_bf16T(const float* __restrict__ w, u16* __restrict__ wt) {
  int i = blockIdx.x * blockDim.x + threadIdx.x;
  if (i >= C_DIM * C_DIM) return;
  int n = i / C_DIM, k = i % C_DIM;            // coalesced writes along k
  wt[i] = f2bf(w[k * C_DIM + n]);
}

// ---------------- kernel 2a: zero head-dim pad of Q/K (d = 56..63) ----------------
__global__ void k_zero_pad_qk(u16* __restrict__ q, u16* __restrict__ k) {
  int i = blockIdx.x * blockDim.x + threadIdx.x;
  if (i >= 2 * BHT) return;
  u16* buf = (i < BHT) ? q : k;
  int row = i % BHT;
  uint4 z = {0u, 0u, 0u, 0u};
  *(uint4*)(buf + (size_t)row * D_PAD + D_HEAD) = z;
}

// ---------------- kernel 2b: zero pad rows of V^T (d = 56..63, all s) -------------
__global__ void k_zero_pad_vt(u16* __restrict__ vt) {
  int i = blockIdx.x * blockDim.x + threadIdx.x;   // over 32*8*2048/8 uint4 chunks
  if (i >= B_DIM * H_DIM * 8 * (T_DIM / 8)) return;
  int bh = i / (8 * (T_DIM / 8));
  int rem = i % (8 * (T_DIM / 8));
  int d = D_HEAD + rem / (T_DIM / 8);
  int c = (rem % (T_DIM / 8)) * 8;
  uint4 z = {0u, 0u, 0u, 0u};
  *(uint4*)(vt + ((size_t)bh * D_PAD + d) * T_DIM + c) = z;
}

// ---------------- kernel 3: QKV projection GEMM ----------------
// A = x_bf16 [8192 x 448]; B = W^T tiles (K-contiguous). Block = 256 thr (8 waves),
// tile 128(M) x 64(N), BK = 32, each wave a 32x32 accumulator (2x2 WMMA tiles).
// z=0 -> Q [B,H,T,64] (scaled), z=1 -> K [B,H,T,64], z=2 -> V^T [B,H,64,T].
__global__ __launch_bounds__(256) void k_gemm_qkv(
    const u16* __restrict__ xb,
    const u16* __restrict__ wqT, const u16* __restrict__ wkT, const u16* __restrict__ wvT,
    u16* __restrict__ qb, u16* __restrict__ kb, u16* __restrict__ vt) {
  __shared__ u16 As[128 * 32];   // 8 KB, [m][k]
  __shared__ u16 Bst[64 * 32];   // 4 KB, [n][k]  (K-contiguous)
  int zz = blockIdx.z;
  const u16* WT = (zz == 0) ? wqT : ((zz == 1) ? wkT : wvT);

  int t = threadIdx.x;
  int m0 = blockIdx.x * 128, n0 = blockIdx.y * 64;
  int w = t >> 5, lane = t & 31, ln = lane & 15, hi = lane >> 4;
  int wm = w >> 1, wn = w & 1;

  v8f z8 = {0.f, 0.f, 0.f, 0.f, 0.f, 0.f, 0.f, 0.f};
  v8f acc[2][2] = {{z8, z8}, {z8, z8}};

  for (int kt = 0; kt < C_DIM / 32; ++kt) {
    {
      int ar = t >> 2, ac = (t & 3) * 8;      // A: 128x32, 2 chunks of 16B
#pragma unroll
      for (int p = 0; p < 2; ++p) {
        int r = ar + p * 64;
        cp16(xb + (size_t)(m0 + r) * C_DIM + kt * 32 + ac, &As[r * 32 + ac]);
      }
      int bn = t >> 2, bc = (t & 3) * 8;      // B^T: 64x32, 1 chunk of 16B
      cp16(WT + (size_t)(n0 + bn) * C_DIM + kt * 32 + bc, &Bst[bn * 32 + bc]);
    }
    cp_wait();
    __syncthreads();
    v16bf a0 = load_fragA_rm(&As[(wm * 32 + 0) * 32], 32, 0);
    v16bf a1 = load_fragA_rm(&As[(wm * 32 + 16) * 32], 32, 0);
    v16bf b0 = load_fragB_kcontig(&Bst[(wn * 32 + 0) * 32], 32, 0);
    v16bf b1 = load_fragB_kcontig(&Bst[(wn * 32 + 16) * 32], 32, 0);
    acc[0][0] = wmma_bf16(a0, b0, acc[0][0]);
    acc[0][1] = wmma_bf16(a0, b1, acc[0][1]);
    acc[1][0] = wmma_bf16(a1, b0, acc[1][0]);
    acc[1][1] = wmma_bf16(a1, b1, acc[1][1]);
    __syncthreads();
  }

#pragma unroll
  for (int mt = 0; mt < 2; ++mt)
#pragma unroll
    for (int nt = 0; nt < 2; ++nt)
#pragma unroll
      for (int r = 0; r < 8; ++r) {
        int gm = m0 + wm * 32 + mt * 16 + r + hi * 8;
        int gn = n0 + wn * 32 + nt * 16 + ln;
        int bb = gm >> 11, tt = gm & (T_DIM - 1);
        int hh = gn / D_HEAD, dp = gn % D_HEAD;
        float val = acc[mt][nt][r];
        if (zz == 2) {
          vt[((size_t)(bb * H_DIM + hh) * D_PAD + dp) * T_DIM + tt] = f2bf(val);
        } else {
          u16* dst = (zz == 0) ? qb : kb;
          if (zz == 0) val *= SCALE_Q;   // fold logit scale into Q
          dst[(((size_t)bb * H_DIM + hh) * T_DIM + tt) * D_PAD + dp] = f2bf(val);
        }
      }
}

// ---------------- kernel 4: flash attention ----------------
// grid = (T/128, H, B); block = 256 (8 waves); wave w owns rows q0 + 16*w so the
// online softmax stays within each wave. K and V^T tiles are staged in LDS
// cooperatively (async-to-LDS), shared by all 8 waves.
__global__ __launch_bounds__(256) void k_attn(
    const u16* __restrict__ qb, const u16* __restrict__ kb,
    const u16* __restrict__ vt, u16* __restrict__ ab) {
  __shared__ u16 Kl[64 * 64];       // 8 KB, [s][d]
  __shared__ u16 Vl[64 * 64];       // 8 KB, [d][s]
  __shared__ u16 Pl[8 * 16 * 64];   // 16 KB, wave-private P tiles

  int t = threadIdx.x, w = t >> 5, lane = t & 31, ln = lane & 15, hi = lane >> 4;
  int bh = blockIdx.z * H_DIM + blockIdx.y;
  const u16* qh = qb + (size_t)bh * T_DIM * D_PAD;
  const u16* kh = kb + (size_t)bh * T_DIM * D_PAD;
  const u16* vh = vt + (size_t)bh * D_PAD * T_DIM;   // [64 d][2048 s]
  int r0 = blockIdx.x * 128 + w * 16;

  v16bf aq0 = load_fragA_rm(qh + (size_t)r0 * D_PAD, D_PAD, 0);
  v16bf aq1 = load_fragA_rm(qh + (size_t)r0 * D_PAD, D_PAD, 32);

  v8f z8 = {0.f, 0.f, 0.f, 0.f, 0.f, 0.f, 0.f, 0.f};
  v8f accO[4] = {z8, z8, z8, z8};
  float rm[8], rl[8];
#pragma unroll
  for (int r = 0; r < 8; ++r) { rm[r] = -__builtin_inff(); rl[r] = 0.f; }

  u16* Pw = &Pl[w * 16 * 64];
  int numS = blockIdx.x * 2 + 2;                 // causal tile range
  for (int st = 0; st < numS; ++st) {
    int s0 = st * 64;
    // ---- stage K tile [s][d] and V^T tile [d][s] into LDS (512 x 16B) ----
#pragma unroll
    for (int p = 0; p < 2; ++p) {
      int i = t + p * 256;
      int row = i >> 3, col = (i & 7) * 8;
      cp16(kh + (size_t)(s0 + row) * D_PAD + col, &Kl[row * D_PAD + col]);
      cp16(vh + (size_t)row * T_DIM + s0 + col, &Vl[row * D_PAD + col]);
    }
    cp_wait();
    __syncthreads();

    // ---- S = Q K^T for this wave's 16 rows x 64 keys (Q pre-scaled) ----
    v8f sac[4];
#pragma unroll
    for (int nt = 0; nt < 4; ++nt) {
      v16bf b0 = load_fragB_kcontig(&Kl[(nt * 16) * D_PAD], D_PAD, 0);
      v16bf b1 = load_fragB_kcontig(&Kl[(nt * 16) * D_PAD], D_PAD, 32);
      sac[nt] = wmma_bf16(aq0, b0, z8);
      sac[nt] = wmma_bf16(aq1, b1, sac[nt]);
    }
    // ---- causal mask on f32 accumulator ----
#pragma unroll
    for (int nt = 0; nt < 4; ++nt)
#pragma unroll
      for (int r = 0; r < 8; ++r) {
        int col = s0 + nt * 16 + ln;
        int row = r0 + r + hi * 8;
        sac[nt][r] = (col <= row) ? sac[nt][r] : -__builtin_inff();
      }
    // ---- online softmax stats (row reductions in 16-lane halves) ----
    float fac[8];
#pragma unroll
    for (int r = 0; r < 8; ++r) {
      float mx = fmaxf(fmaxf(sac[0][r], sac[1][r]), fmaxf(sac[2][r], sac[3][r]));
      mx = fmaxf(mx, __shfl_xor(mx, 1, 32));
      mx = fmaxf(mx, __shfl_xor(mx, 2, 32));
      mx = fmaxf(mx, __shfl_xor(mx, 4, 32));
      mx = fmaxf(mx, __shfl_xor(mx, 8, 32));
      float mn = fmaxf(rm[r], mx);
      fac[r] = __expf(rm[r] - mn);
      rm[r] = mn;
    }
    float rs[8];
#pragma unroll
    for (int r = 0; r < 8; ++r) rs[r] = 0.f;
#pragma unroll
    for (int nt = 0; nt < 4; ++nt)
#pragma unroll
      for (int r = 0; r < 8; ++r) {
        float p = __expf(sac[nt][r] - rm[r]);
        rs[r] += p;
        Pw[(r + hi * 8) * D_PAD + nt * 16 + ln] = f2bf(p);  // C-layout -> LDS
      }
#pragma unroll
    for (int r = 0; r < 8; ++r) {
      float s = rs[r];
      s += __shfl_xor(s, 1, 32);
      s += __shfl_xor(s, 2, 32);
      s += __shfl_xor(s, 4, 32);
      s += __shfl_xor(s, 8, 32);
      rl[r] = rl[r] * fac[r] + s;
#pragma unroll
      for (int nt2 = 0; nt2 < 4; ++nt2)
        accO[nt2][r] = accO[nt2][r] * fac[r];
    }
    // ---- O += P @ V  (P back through LDS as A-layout; V^T tile K-contiguous) ----
    v16bf ap0 = load_fragA_rm(Pw, D_PAD, 0);
    v16bf ap1 = load_fragA_rm(Pw, D_PAD, 32);
#pragma unroll
    for (int nt2 = 0; nt2 < 4; ++nt2) {
      v16bf bv0 = load_fragB_kcontig(&Vl[(nt2 * 16) * D_PAD], D_PAD, 0);
      v16bf bv1 = load_fragB_kcontig(&Vl[(nt2 * 16) * D_PAD], D_PAD, 32);
      accO[nt2] = wmma_bf16(ap0, bv0, accO[nt2]);
      accO[nt2] = wmma_bf16(ap1, bv1, accO[nt2]);
    }
    __syncthreads();   // protect Kl/Vl before next tile's staging
  }
  // ---- normalize and write [B,T,H*56] bf16 ----
#pragma unroll
  for (int r = 0; r < 8; ++r) {
    float inv = 1.0f / rl[r];
    int row = r0 + r + hi * 8;
    size_t gm = (size_t)blockIdx.z * T_DIM + row;
#pragma unroll
    for (int nt2 = 0; nt2 < 4; ++nt2) {
      int d = nt2 * 16 + ln;
      if (d < D_HEAD)
        ab[gm * C_DIM + blockIdx.y * D_HEAD + d] = f2bf(accO[nt2][r] * inv);
    }
  }
}

// ---------------- kernel 5: output projection GEMM (+bias, f32 out) ----------------
__global__ __launch_bounds__(256) void k_gemm_proj(
    const u16* __restrict__ ab, const u16* __restrict__ wpT,
    const float* __restrict__ bp, float* __restrict__ out) {
  __shared__ u16 As[128 * 32];
  __shared__ u16 Bst[64 * 32];
  int t = threadIdx.x;
  int m0 = blockIdx.x * 128, n0 = blockIdx.y * 64;
  int w = t >> 5, lane = t & 31, ln = lane & 15, hi = lane >> 4;
  int wm = w >> 1, wn = w & 1;

  v8f z8 = {0.f, 0.f, 0.f, 0.f, 0.f, 0.f, 0.f, 0.f};
  v8f acc[2][2] = {{z8, z8}, {z8, z8}};

  for (int kt = 0; kt < C_DIM / 32; ++kt) {
    {
      int ar = t >> 2, ac = (t & 3) * 8;
#pragma unroll
      for (int p = 0; p < 2; ++p) {
        int r = ar + p * 64;
        cp16(ab + (size_t)(m0 + r) * C_DIM + kt * 32 + ac, &As[r * 32 + ac]);
      }
      int bn = t >> 2, bc = (t & 3) * 8;
      cp16(wpT + (size_t)(n0 + bn) * C_DIM + kt * 32 + bc, &Bst[bn * 32 + bc]);
    }
    cp_wait();
    __syncthreads();
    v16bf a0 = load_fragA_rm(&As[(wm * 32 + 0) * 32], 32, 0);
    v16bf a1 = load_fragA_rm(&As[(wm * 32 + 16) * 32], 32, 0);
    v16bf b0 = load_fragB_kcontig(&Bst[(wn * 32 + 0) * 32], 32, 0);
    v16bf b1 = load_fragB_kcontig(&Bst[(wn * 32 + 16) * 32], 32, 0);
    acc[0][0] = wmma_bf16(a0, b0, acc[0][0]);
    acc[0][1] = wmma_bf16(a0, b1, acc[0][1]);
    acc[1][0] = wmma_bf16(a1, b0, acc[1][0]);
    acc[1][1] = wmma_bf16(a1, b1, acc[1][1]);
    __syncthreads();
  }

#pragma unroll
  for (int mt = 0; mt < 2; ++mt)
#pragma unroll
    for (int nt = 0; nt < 2; ++nt)
#pragma unroll
      for (int r = 0; r < 8; ++r) {
        int gm = m0 + wm * 32 + mt * 16 + r + hi * 8;
        int gn = n0 + wn * 32 + nt * 16 + ln;
        out[(size_t)gm * C_DIM + gn] = acc[mt][nt][r] + bp[gn];
      }
}

// ---------------- host-side launch ----------------
extern "C" void kernel_launch(void* const* d_in, const int* in_sizes, int n_in,
                              void* d_out, int out_size, void* d_ws, size_t ws_size,
                              hipStream_t stream) {
  (void)in_sizes; (void)n_in; (void)out_size; (void)ws_size;
  const float* x  = (const float*)d_in[0];
  const float* Wq = (const float*)d_in[1];
  const float* Wk = (const float*)d_in[2];
  const float* Wv = (const float*)d_in[3];
  const float* Wp = (const float*)d_in[4];
  const float* bp = (const float*)d_in[5];
  float* out = (float*)d_out;

  char* ws = (char*)d_ws;
  size_t off = 0;
  auto give = [&](size_t bytes) -> char* {
    char* p = ws + off;
    off += (bytes + 255) & ~(size_t)255;
    return p;
  };
  u16* xb  = (u16*)give((size_t)BT * C_DIM * 2);       // x in bf16
  u16* wqT = (u16*)give((size_t)C_DIM * C_DIM * 2);    // W^T bf16 [N][K]
  u16* wkT = (u16*)give((size_t)C_DIM * C_DIM * 2);
  u16* wvT = (u16*)give((size_t)C_DIM * C_DIM * 2);
  u16* wpT = (u16*)give((size_t)C_DIM * C_DIM * 2);
  u16* qb  = (u16*)give((size_t)BHT * D_PAD * 2);      // Q [B,H,T,64] (pre-scaled)
  u16* kb  = (u16*)give((size_t)BHT * D_PAD * 2);      // K [B,H,T,64]
  u16* vt  = (u16*)give((size_t)BHT * D_PAD * 2);      // V^T [B,H,64,T]
  u16* ab  = (u16*)give((size_t)BT * C_DIM * 2);       // attention out bf16

  k_f32_to_bf16<<<2048, 256, 0, stream>>>(x, xb, BT * C_DIM);
  k_w_to_bf16T<<<784, 256, 0, stream>>>(Wq, wqT);
  k_w_to_bf16T<<<784, 256, 0, stream>>>(Wk, wkT);
  k_w_to_bf16T<<<784, 256, 0, stream>>>(Wv, wvT);
  k_w_to_bf16T<<<784, 256, 0, stream>>>(Wp, wpT);
  k_zero_pad_qk<<<(2 * BHT) / 256, 256, 0, stream>>>(qb, kb);
  k_zero_pad_vt<<<(B_DIM * H_DIM * 8 * (T_DIM / 8)) / 256, 256, 0, stream>>>(vt);

  k_gemm_qkv<<<dim3(BT / 128, C_DIM / 64, 3), 256, 0, stream>>>(
      xb, wqT, wkT, wvT, qb, kb, vt);
  k_attn<<<dim3(T_DIM / 128, H_DIM, B_DIM), 256, 0, stream>>>(qb, kb, vt, ab);
  k_gemm_proj<<<dim3(BT / 128, C_DIM / 64, 1), 256, 0, stream>>>(ab, wpT, bp, out);
}
